// MultiHeadAttention_18700287607660
// MI455X (gfx1250) — matrix operand order repaired and checked by reading, not compiled, not run on hardware
//
#include <hip/hip_runtime.h>
#include <hip/hip_bf16.h>

// ---------------------------------------------------------------------------
// B=128, Lq=Lk=500, D=512, NWAY=5.  Algebraic folding (mean commutes with
// linear maps + mean-before-softmax):
//   qbar_in = group-mean(q)            [B,5,D]   (bf16)
//   qbar    = qbar_in @ Wq^T           tiny WMMA GEMM
//   kp      = k @ Wk^T                 big WMMA GEMM (bf16 in, fp32 out)
//   attn    = softmax(qbar.kp/sqrt(D)) [B,5,L]
//   av      = attn @ v                 [B,5,D]   (bf16)
//   proto   = av @ Wv^T                tiny WMMA GEMM
//   h       = LN1(proto_bcast + kp)    (fp32 + bf16 copy)
//   f       = h @ Wfc^T                big WMMA GEMM
//   out     = LN2(leaky(f) + h)
// All GEMMs: bf16 A/B via v_wmma_f32_16x16x32_bf16, fp32 accumulate.
// Tiles are staged into LDS with global_load_async_to_lds_b128 (ASYNCcnt),
// double-buffered.
// ---------------------------------------------------------------------------

#define D_MODEL 512
#define L_SEQ   500
#define BATCH   128
#define NWAY    5
#define NGRP    100
#define M_BIG   (BATCH * L_SEQ)        // 64000
#define M_SMALL (BATCH * NWAY)         // 640
#define TEMP_INV (1.0f / 22.62741699796952f)   // 1/sqrt(512)
#define LEAKY   0.1f

typedef __bf16 bf16_t;
typedef bf16_t v16bf __attribute__((ext_vector_type(16)));
typedef bf16_t v8bf  __attribute__((ext_vector_type(8)));
typedef float  v8f   __attribute__((ext_vector_type(8)));

// Issue one 16-byte async global->LDS DMA (tracked by ASYNCcnt).
__device__ __forceinline__ void async_ld16(unsigned lds_off, const void* g) {
  unsigned long long ga = (unsigned long long)(uintptr_t)g;
  asm volatile("global_load_async_to_lds_b128 %0, %1, off"
               :: "v"(lds_off), "v"(ga) : "memory");
}
__device__ __forceinline__ void wait_async0() {
  asm volatile("s_wait_asynccnt 0x0" ::: "memory");
}

// ---------------------------------------------------------------------------
// Y[M,512] = X[M,512] @ W[512,512]^T, X/W bf16, Y fp32.
// Block 256 thr (8 waves), tile 128(M) x 64(N), K step 32, LDS double-buffered
// via async DMA.  Wave grid 4x2, each wave 32x32 = 2x2 WMMA accumulators.
// ---------------------------------------------------------------------------
__global__ __launch_bounds__(256)
void gemm_bf16_async(const bf16_t* __restrict__ X, const bf16_t* __restrict__ W,
                     float* __restrict__ Y) {
  constexpr int K = 512, N = 512;
  constexpr int LDT = 40;                          // padded row stride (80 B)
  __shared__ alignas(16) bf16_t As[2][128 * LDT];  // 2 x 10240 B
  __shared__ alignas(16) bf16_t Ws[2][64 * LDT];   // 2 x  5120 B

  const int tid  = threadIdx.x;
  const int lane = tid & 31;
  const int wave = tid >> 5;
  const int wm   = (wave & 3) * 32;
  const int wn   = (wave >> 2) * 32;
  const int m0   = blockIdx.y * 128;
  const int n0   = blockIdx.x * 64;
  const int half = lane >> 4;
  const int l16  = lane & 15;

  // Staging decomposition: 16B chunk = 8 bf16.  A: 128 rows x 4 chunks
  // (2 per thread), W: 64 rows x 4 chunks (1 per thread).
  const int ar = tid >> 2;            // 0..63
  const int ac = (tid & 3) * 8;       // element col within 32-wide tile
  const unsigned aldsA0 = (unsigned)(uintptr_t)&As[0][0];
  const unsigned aldsA1 = (unsigned)(uintptr_t)&As[1][0];
  const unsigned aldsW0 = (unsigned)(uintptr_t)&Ws[0][0];
  const unsigned aldsW1 = (unsigned)(uintptr_t)&Ws[1][0];

  auto stage = [&](int p, int kk) {
    unsigned ab = p ? aldsA1 : aldsA0;
    unsigned wb = p ? aldsW1 : aldsW0;
    async_ld16(ab + (unsigned)((ar        * LDT + ac) * 2),
               X + (size_t)(m0 + ar)      * K + kk + ac);
    async_ld16(ab + (unsigned)(((ar + 64) * LDT + ac) * 2),
               X + (size_t)(m0 + ar + 64) * K + kk + ac);
    async_ld16(wb + (unsigned)((ar        * LDT + ac) * 2),
               W + (size_t)(n0 + ar)      * K + kk + ac);
  };

  v8f acc[2][2] = {};
  stage(0, 0);

  for (int kk = 0; kk < K; kk += 32) {
    const int p = (kk >> 5) & 1;
    wait_async0();        // our wave's DMAs into buffer p have landed
    __syncthreads();      // every wave's DMAs landed; prior reads of p^1 done
    if (kk + 32 < K) stage(p ^ 1, kk + 32);

    const bf16_t* Ab = &As[p][0];
    const bf16_t* Wb = &Ws[p][0];

    v16bf a[2], b[2];
    #pragma unroll
    for (int i = 0; i < 2; ++i) {
      // A 16x32 layout: lane half h holds K in [8h,8h+8) then [16+8h,16+8h+8)
      const bf16_t* base = Ab + (wm + i * 16 + l16) * LDT + half * 8;
      v8bf lo = *reinterpret_cast<const v8bf*>(base);
      v8bf hi = *reinterpret_cast<const v8bf*>(base + 16);
      #pragma unroll
      for (int e = 0; e < 8; ++e) { a[i][e] = lo[e]; a[i][e + 8] = hi[e]; }
    }
    #pragma unroll
    for (int j = 0; j < 2; ++j) {
      // B 32x16 layout: lane half h holds K in [8h,8h+8) then [16+8h,16+8h+8)
      const bf16_t* base = Wb + (wn + j * 16 + l16) * LDT + half * 8;
      v8bf lo = *reinterpret_cast<const v8bf*>(base);
      v8bf hi = *reinterpret_cast<const v8bf*>(base + 16);
      #pragma unroll
      for (int e = 0; e < 8; ++e) { b[j][e] = lo[e]; b[j][e + 8] = hi[e]; }
    }
    #pragma unroll
    for (int i = 0; i < 2; ++i)
      #pragma unroll
      for (int j = 0; j < 2; ++j)
        acc[i][j] = __builtin_amdgcn_wmma_f32_16x16x32_bf16(
            false, a[i], false, b[j], (short)0, acc[i][j], false, false);
    __syncthreads();      // reads of buffer p complete before it is re-filled
  }

  // C layout: VGPR r -> M = r + 8*(lane/16), N = lane%16
  #pragma unroll
  for (int i = 0; i < 2; ++i) {
    int rbase = m0 + wm + i * 16 + half * 8;
    #pragma unroll
    for (int j = 0; j < 2; ++j) {
      int col = n0 + wn + j * 16 + l16;
      #pragma unroll
      for (int r = 0; r < 8; ++r)
        Y[(size_t)(rbase + r) * N + col] = acc[i][j][r];
    }
  }
}

// ---------------------------------------------------------------------------
// Streaming fp32 -> bf16 conversion, 8 elems/thread (packed v_cvt path).
// ---------------------------------------------------------------------------
__global__ __launch_bounds__(256)
void cvt_bf16_kernel(const float* __restrict__ src, bf16_t* __restrict__ dst,
                     int n8) {
  int i = blockIdx.x * 256 + threadIdx.x;
  if (i >= n8) return;
  v8f x = *reinterpret_cast<const v8f*>(src + (size_t)i * 8);
  v8bf o = __builtin_convertvector(x, v8bf);
  *reinterpret_cast<v8bf*>(dst + (size_t)i * 8) = o;
}

// ---------------------------------------------------------------------------
// qbar_in[b,w,d] = mean_j q[b, j*5+w, d]   (bf16 output for GEMM)
// ---------------------------------------------------------------------------
__global__ __launch_bounds__(256)
void reduce_q_kernel(const float* __restrict__ q, bf16_t* __restrict__ qbar_in) {
  int idx = blockIdx.x * 256 + threadIdx.x;     // 640*512 total
  int d  = idx & 511;
  int bw = idx >> 9;
  int b = bw / NWAY, w = bw % NWAY;
  const float* base = q + ((size_t)b * L_SEQ + w) * D_MODEL + d;
  float s = 0.f;
  #pragma unroll 4
  for (int j = 0; j < NGRP; ++j) s += base[(size_t)j * NWAY * D_MODEL];
  qbar_in[idx] = (bf16_t)(s * (1.0f / NGRP));
}

// ---------------------------------------------------------------------------
// attn[b,w,k] = softmax_k( qbar[b,w,:] . kp[b,k,:] / sqrt(D) ).  1 block / b.
// ---------------------------------------------------------------------------
__global__ __launch_bounds__(256)
void scores_softmax_kernel(const float* __restrict__ qbar,
                           const float* __restrict__ kp,
                           float* __restrict__ attn) {
  __shared__ float qb[NWAY * 512];
  __shared__ float sc[NWAY * 512];
  const int b = blockIdx.x;
  const int tid = threadIdx.x, lane = tid & 31, wave = tid >> 5;

  for (int i = tid; i < NWAY * 512; i += 256)
    qb[i] = qbar[(size_t)b * NWAY * 512 + i];
  __syncthreads();

  for (int k = wave; k < L_SEQ; k += 8) {
    const float* kr = kp + ((size_t)b * L_SEQ + k) * 512;
    float acc[NWAY] = {0.f, 0.f, 0.f, 0.f, 0.f};
    #pragma unroll
    for (int u = 0; u < 16; u += 4) {
      float4 kv = *reinterpret_cast<const float4*>(kr + lane * 16 + u);
      #pragma unroll
      for (int w = 0; w < NWAY; ++w) {
        float4 qv = *reinterpret_cast<const float4*>(&qb[w * 512 + lane * 16 + u]);
        acc[w] += kv.x*qv.x + kv.y*qv.y + kv.z*qv.z + kv.w*qv.w;
      }
    }
    #pragma unroll
    for (int w = 0; w < NWAY; ++w) {
      float s = acc[w];
      for (int m = 16; m > 0; m >>= 1) s += __shfl_xor(s, m, 32);
      if (lane == 0) sc[w * 512 + k] = s * TEMP_INV;
    }
  }
  __syncthreads();

  if (wave < NWAY) {
    float mx = -3.4e38f;
    for (int k = lane; k < L_SEQ; k += 32) mx = fmaxf(mx, sc[wave * 512 + k]);
    for (int m = 16; m > 0; m >>= 1) mx = fmaxf(mx, __shfl_xor(mx, m, 32));
    float sum = 0.f;
    for (int k = lane; k < L_SEQ; k += 32) {
      float e = __expf(sc[wave * 512 + k] - mx);
      sc[wave * 512 + k] = e;
      sum += e;
    }
    for (int m = 16; m > 0; m >>= 1) sum += __shfl_xor(sum, m, 32);
    float inv = 1.0f / sum;
    for (int k = lane; k < L_SEQ; k += 32)
      attn[((size_t)b * NWAY + wave) * L_SEQ + k] = sc[wave * 512 + k] * inv;
  }
}

// ---------------------------------------------------------------------------
// av[b,w,e] = sum_k attn[b,w,k] * v[b,k,e]  (bf16 output).  1 block / b.
// ---------------------------------------------------------------------------
__global__ __launch_bounds__(256)
void av_kernel(const float* __restrict__ attn, const float* __restrict__ v,
               bf16_t* __restrict__ av) {
  __shared__ float sa[NWAY * L_SEQ];
  const int b = blockIdx.x, tid = threadIdx.x;
  for (int i = tid; i < NWAY * L_SEQ; i += 256)
    sa[i] = attn[(size_t)b * NWAY * L_SEQ + i];
  __syncthreads();
  float acc[NWAY][2] = {};
  for (int k = 0; k < L_SEQ; ++k) {
    const float* vr = v + ((size_t)b * L_SEQ + k) * 512;
    float v0 = vr[tid], v1 = vr[tid + 256];
    #pragma unroll
    for (int w = 0; w < NWAY; ++w) {
      float a = sa[w * L_SEQ + k];
      acc[w][0] = fmaf(a, v0, acc[w][0]);
      acc[w][1] = fmaf(a, v1, acc[w][1]);
    }
  }
  #pragma unroll
  for (int w = 0; w < NWAY; ++w) {
    av[((size_t)b * NWAY + w) * 512 + tid]       = (bf16_t)acc[w][0];
    av[((size_t)b * NWAY + w) * 512 + tid + 256] = (bf16_t)acc[w][1];
  }
}

// ---------------------------------------------------------------------------
// h = LN1(proto_bcast + kp)*g1 + b1 ; writes fp32 h and bf16 hbf.  Wave/row.
// ---------------------------------------------------------------------------
__global__ __launch_bounds__(256)
void ln1_kernel(const float* __restrict__ kp, const float* __restrict__ proto,
                const float* __restrict__ g1, const float* __restrict__ b1,
                float* __restrict__ h, bf16_t* __restrict__ hbf) {
  const int lane = threadIdx.x & 31, wave = threadIdx.x >> 5;
  const int row = blockIdx.x * 8 + wave;
  const int b = row / L_SEQ, l = row % L_SEQ, w = l % NWAY;
  const float* kr = kp + (size_t)row * 512;
  const float* pr = proto + ((size_t)b * NWAY + w) * 512;
  float x[16]; float sum = 0.f, sq = 0.f;
  #pragma unroll
  for (int u = 0; u < 16; ++u) {
    int d = lane + u * 32;
    float t = kr[d] + pr[d];
    x[u] = t; sum += t; sq += t * t;
  }
  for (int m = 16; m > 0; m >>= 1) {
    sum += __shfl_xor(sum, m, 32);
    sq  += __shfl_xor(sq,  m, 32);
  }
  float mean = sum * (1.0f / 512.0f);
  float var  = sq  * (1.0f / 512.0f) - mean * mean;
  float rstd = rsqrtf(var + 1e-6f);
  #pragma unroll
  for (int u = 0; u < 16; ++u) {
    int d = lane + u * 32;
    float o = (x[u] - mean) * rstd * g1[d] + b1[d];
    h[(size_t)row * 512 + d]   = o;
    hbf[(size_t)row * 512 + d] = (bf16_t)o;
  }
}

// ---------------------------------------------------------------------------
// out = LN2(leaky(f) + h)*g2 + b2.  Wave per row.
// ---------------------------------------------------------------------------
__global__ __launch_bounds__(256)
void ln2_kernel(const float* __restrict__ f, const float* __restrict__ h,
                const float* __restrict__ g2, const float* __restrict__ b2,
                float* __restrict__ out) {
  const int lane = threadIdx.x & 31, wave = threadIdx.x >> 5;
  const int row = blockIdx.x * 8 + wave;
  const float* fr = f + (size_t)row * 512;
  const float* hr = h + (size_t)row * 512;
  float x[16]; float sum = 0.f, sq = 0.f;
  #pragma unroll
  for (int u = 0; u < 16; ++u) {
    int d = lane + u * 32;
    float fv = fr[d];
    float t = (fv > 0.f ? fv : LEAKY * fv) + hr[d];
    x[u] = t; sum += t; sq += t * t;
  }
  for (int m = 16; m > 0; m >>= 1) {
    sum += __shfl_xor(sum, m, 32);
    sq  += __shfl_xor(sq,  m, 32);
  }
  float mean = sum * (1.0f / 512.0f);
  float var  = sq  * (1.0f / 512.0f) - mean * mean;
  float rstd = rsqrtf(var + 1e-6f);
  #pragma unroll
  for (int u = 0; u < 16; ++u) {
    int d = lane + u * 32;
    out[(size_t)row * 512 + d] = (x[u] - mean) * rstd * g2[d] + b2[d];
  }
}

// ---------------------------------------------------------------------------
extern "C" void kernel_launch(void* const* d_in, const int* in_sizes, int n_in,
                              void* d_out, int out_size, void* d_ws, size_t ws_size,
                              hipStream_t stream) {
  (void)in_sizes; (void)n_in; (void)out_size; (void)ws_size;
  const float* q   = (const float*)d_in[0];
  const float* k   = (const float*)d_in[1];
  const float* v   = (const float*)d_in[2];
  const float* Wq  = (const float*)d_in[3];
  const float* Wk  = (const float*)d_in[4];
  const float* Wv  = (const float*)d_in[5];
  const float* Wfc = (const float*)d_in[6];
  const float* g1  = (const float*)d_in[7];
  const float* b1  = (const float*)d_in[8];
  const float* g2  = (const float*)d_in[9];
  const float* b2  = (const float*)d_in[10];
  float* out = (float*)d_out;

  // Workspace layout (all region sizes multiples of 4 floats -> 16B aligned).
  float* ws = (float*)d_ws;
  size_t o = 0;
  float*  kp_f    = ws + o; o += (size_t)M_BIG * 512;        // kp; reused as fc out
  float*  h_buf   = ws + o; o += (size_t)M_BIG * 512;        // fp32 h
  bf16_t* bfbuf   = (bf16_t*)(ws + o); o += (size_t)M_BIG * 256;  // kbf, later hbf
  bf16_t* wqbf    = (bf16_t*)(ws + o); o += 512 * 256;
  bf16_t* wkbf    = (bf16_t*)(ws + o); o += 512 * 256;
  bf16_t* wvbf    = (bf16_t*)(ws + o); o += 512 * 256;
  bf16_t* wfcbf   = (bf16_t*)(ws + o); o += 512 * 256;
  float*  qbar    = ws + o; o += (size_t)M_SMALL * 512;
  bf16_t* qbarinb = (bf16_t*)(ws + o); o += (size_t)M_SMALL * 256;
  float*  attn    = ws + o; o += (size_t)M_SMALL * L_SEQ;
  bf16_t* av_bf   = (bf16_t*)(ws + o); o += (size_t)M_SMALL * 256;
  float*  proto   = ws + o; o += (size_t)M_SMALL * 512;
  // total ~334 MB

  const int n8_big = (M_BIG * 512) / 8;       // 4,096,000
  const int n8_w   = (512 * 512) / 8;         // 32,768

  cvt_bf16_kernel<<<(n8_big + 255) / 256, 256, 0, stream>>>(k,   bfbuf, n8_big);
  cvt_bf16_kernel<<<(n8_w   + 255) / 256, 256, 0, stream>>>(Wq,  wqbf,  n8_w);
  cvt_bf16_kernel<<<(n8_w   + 255) / 256, 256, 0, stream>>>(Wk,  wkbf,  n8_w);
  cvt_bf16_kernel<<<(n8_w   + 255) / 256, 256, 0, stream>>>(Wv,  wvbf,  n8_w);
  cvt_bf16_kernel<<<(n8_w   + 255) / 256, 256, 0, stream>>>(Wfc, wfcbf, n8_w);

  reduce_q_kernel<<<(M_SMALL * 512) / 256, 256, 0, stream>>>(q, qbarinb);
  gemm_bf16_async<<<dim3(8, M_SMALL / 128), 256, 0, stream>>>(qbarinb, wqbf, qbar);
  gemm_bf16_async<<<dim3(8, M_BIG / 128),   256, 0, stream>>>(bfbuf,   wkbf, kp_f);
  scores_softmax_kernel<<<BATCH, 256, 0, stream>>>(qbar, kp_f, attn);
  av_kernel<<<BATCH, 256, 0, stream>>>(attn, v, av_bf);
  gemm_bf16_async<<<dim3(8, M_SMALL / 128), 256, 0, stream>>>(av_bf, wvbf, proto);
  ln1_kernel<<<M_BIG / 8, 256, 0, stream>>>(kp_f, proto, g1, b1, h_buf, bfbuf);
  gemm_bf16_async<<<dim3(8, M_BIG / 128),   256, 0, stream>>>(bfbuf, wfcbf, kp_f);
  ln2_kernel<<<M_BIG / 8, 256, 0, stream>>>(kp_f, h_buf, g2, b2, out);
}
